// SpikingQNetwork_53369263620404
// MI455X (gfx1250) — matrix operand order repaired and checked by reading, not compiled, not run on hardware
//
#include <hip/hip_runtime.h>
#include <hip/hip_bf16.h>

typedef __attribute__((ext_vector_type(16))) _Float16 v16h;
typedef __attribute__((ext_vector_type(8)))  float    v8f;

// Problem constants (from reference setup_inputs)
constexpr int BATCH = 4096;
constexpr int OBS   = 256;
constexpr int H1    = 1024;
constexpr int H2    = 512;
constexpr int ACT   = 4;
constexpr int TSTEP = 8;

// ---------------------------------------------------------------------------
// Fragment loaders (wave32 WMMA 16x16x32 f16 layouts, CDNA5 ISA 7.12.2)
// A-matrix 16x32: lane L -> row M=L%16; halves 0..7 = K[hi*8 .. hi*8+7],
//                 halves 8..15 = K[16+hi*8 .. 16+hi*8+7]   (hi = L>>4)
// B-matrix 32x16: lane L -> col N=L%16; same K order as A (hi*8.., 16+hi*8..)
// Both read K-contiguous 16B chunks from row-major [row][k] f16 storage.
// ---------------------------------------------------------------------------
__device__ __forceinline__ v16h load_frag_a(const _Float16* __restrict__ p, int ldk,
                                            int m0, int k0, int r, int hi) {
  const _Float16* row = p + (size_t)(m0 + r) * ldk + k0 + hi * 8;
  union { uint4 u[2]; v16h h; } t;
  t.u[0] = *reinterpret_cast<const uint4*>(row);
  t.u[1] = *reinterpret_cast<const uint4*>(row + 16);
  return t.h;
}

__device__ __forceinline__ v16h load_frag_b(const _Float16* __restrict__ p, int ldk,
                                            int n0, int k0, int r, int hi) {
  const _Float16* row = p + (size_t)(n0 + r) * ldk + k0 + hi * 8;
  union { uint4 u[2]; v16h h; } t;
  t.u[0] = *reinterpret_cast<const uint4*>(row);
  t.u[1] = *reinterpret_cast<const uint4*>(row + 16);
  return t.h;
}

__device__ __forceinline__ float clip01(float b) { return fminf(fmaxf(b, 0.0f), 1.0f); }

#define WMMA_F16(A, B, C) \
  __builtin_amdgcn_wmma_f32_16x16x32_f16(false, (A), false, (B), (short)0, (C), false, false)

// ---------------------------------------------------------------------------
// f32 -> f16 conversion (x, W1, W2)
// ---------------------------------------------------------------------------
__global__ void cvt_f32_f16(const float* __restrict__ in, _Float16* __restrict__ out, int n) {
  int i = blockIdx.x * blockDim.x + threadIdx.x;
  if (i < n) out[i] = (_Float16)in[i];
}

// ---------------------------------------------------------------------------
// Kernel A: cur1 = x @ W1^T + b1 (WMMA), then fused 8-step LIF1 in registers.
// Emits spk1[t][b][h] as f16 {0,1}. One wave per 16x16 (b,h) tile.
// ---------------------------------------------------------------------------
__global__ void gemm1_lif1(const _Float16* __restrict__ xh,
                           const _Float16* __restrict__ w1h,
                           const float* __restrict__ b1,
                           const float* __restrict__ beta1p,
                           _Float16* __restrict__ spk1) {
  const int wave = (blockIdx.x * blockDim.x + threadIdx.x) >> 5;
  const int lane = threadIdx.x & 31;
  const int NT = H1 / 16;                 // 64 tiles along h
  if (wave >= (BATCH / 16) * NT) return;
  const int nt = wave % NT, mt = wave / NT;
  const int r = lane & 15, hi = lane >> 4;
  const int m0 = mt * 16, n0 = nt * 16;

  v8f c = {};
  for (int k0 = 0; k0 < OBS; k0 += 32) {
    v16h a = load_frag_a(xh,  OBS, m0, k0, r, hi);
    v16h b = load_frag_b(w1h, OBS, n0, k0, r, hi);
    c = WMMA_F16(a, b, c);
  }

  const float beta = clip01(beta1p[0]);
  const int h = n0 + r;
  const float bias = b1[h];

#pragma unroll
  for (int j = 0; j < 8; ++j) {
    const int brow = m0 + j + hi * 8;     // D layout: VGPR j -> M = j + 8*hi, N = r
    const float cur = c[j] + bias;
    float mem = 0.0f;
    const size_t base = (size_t)brow * H1 + h;
#pragma unroll
    for (int t = 0; t < TSTEP; ++t) {
      const float reset = (mem > 1.0f) ? 1.0f : 0.0f;   // th = 1.0, soft reset
      mem = beta * mem + cur - reset;
      spk1[(size_t)t * ((size_t)BATCH * H1) + base] =
          (mem > 1.0f) ? (_Float16)1.0f : (_Float16)0.0f;
    }
  }
}

// ---------------------------------------------------------------------------
// Kernel B: per step t: cur2 = spk1[t] @ W2^T + b2 (WMMA, K=1024), fused LIF2.
// 2x2 register tiling: one wave owns a 32x32 (b,h2) output block ->
// 4 WMMAs per 4 fragment loads (2x the flops/byte at the L0 interface).
// mem2 state (4 tiles x 8 lanes-worth) persists in VGPRs across the 8 steps.
// ---------------------------------------------------------------------------
__global__ void gemm2_lif2(const _Float16* __restrict__ spk1,
                           const _Float16* __restrict__ w2h,
                           const float* __restrict__ b2,
                           const float* __restrict__ beta2p,
                           _Float16* __restrict__ spk2) {
  const int wave = (blockIdx.x * blockDim.x + threadIdx.x) >> 5;
  const int lane = threadIdx.x & 31;
  const int NT = H2 / 32;                 // 16 super-tiles along h2
  if (wave >= (BATCH / 32) * NT) return;  // 128 * 16 = 2048 wave-tiles
  const int nt = wave % NT, mt = wave / NT;
  const int r = lane & 15, hi = lane >> 4;
  const int m0 = mt * 32, n0 = nt * 32;

  const float beta = clip01(beta2p[0]);
  const float bias[2] = { b2[n0 + r], b2[n0 + 16 + r] };

  float mem[4][8];
#pragma unroll
  for (int q = 0; q < 4; ++q)
#pragma unroll
    for (int j = 0; j < 8; ++j) mem[q][j] = 0.0f;

  for (int t = 0; t < TSTEP; ++t) {
    const _Float16* At = spk1 + (size_t)t * ((size_t)BATCH * H1);
    v8f c00 = {}, c01 = {}, c10 = {}, c11 = {};
    for (int k0 = 0; k0 < H1; k0 += 32) {
      v16h a0 = load_frag_a(At,  H1, m0,      k0, r, hi);
      v16h a1 = load_frag_a(At,  H1, m0 + 16, k0, r, hi);
      v16h b0 = load_frag_b(w2h, H1, n0,      k0, r, hi);
      v16h b1 = load_frag_b(w2h, H1, n0 + 16, k0, r, hi);
      c00 = WMMA_F16(a0, b0, c00);
      c01 = WMMA_F16(a0, b1, c01);
      c10 = WMMA_F16(a1, b0, c10);
      c11 = WMMA_F16(a1, b1, c11);
    }

    _Float16* St = spk2 + (size_t)t * ((size_t)BATCH * H2);
#pragma unroll
    for (int mi = 0; mi < 2; ++mi) {
#pragma unroll
      for (int ni = 0; ni < 2; ++ni) {
        const int q = mi * 2 + ni;
        const v8f& c = (mi == 0) ? (ni == 0 ? c00 : c01) : (ni == 0 ? c10 : c11);
        const int h2 = n0 + ni * 16 + r;
#pragma unroll
        for (int j = 0; j < 8; ++j) {
          const int brow = m0 + mi * 16 + j + hi * 8;
          const float cur = c[j] + bias[ni];
          const float reset = (mem[q][j] > 1.0f) ? 1.0f : 0.0f;
          mem[q][j] = beta * mem[q][j] + cur - reset;
          St[(size_t)brow * H2 + h2] =
              (mem[q][j] > 1.0f) ? (_Float16)1.0f : (_Float16)0.0f;
        }
      }
    }
  }
}

// ---------------------------------------------------------------------------
// Kernel C: output layer. One wave per batch row b.
// Each lane owns 16 h2 elements (two b128 loads); wave-reduce 4 dot products
// per step via __shfl_xor (wave32), then leaky-integrate mem_out, track max.
// ---------------------------------------------------------------------------
__global__ void out_layer(const _Float16* __restrict__ spk2,
                          const float* __restrict__ wout,
                          const float* __restrict__ bout,
                          const float* __restrict__ betaop,
                          float* __restrict__ out) {
  const int wave = (blockIdx.x * blockDim.x + threadIdx.x) >> 5;
  const int lane = threadIdx.x & 31;
  if (wave >= BATCH) return;
  const int b = wave;
  const float beta = clip01(betaop[0]);

  float memo[ACT], mx[ACT];
#pragma unroll
  for (int a = 0; a < ACT; ++a) { memo[a] = 0.0f; mx[a] = -1e9f; }

  for (int t = 0; t < TSTEP; ++t) {
    const _Float16* row = spk2 + (size_t)t * ((size_t)BATCH * H2) + (size_t)b * H2 + lane * 16;
    union { uint4 u[2]; _Float16 h[16]; } d;
    d.u[0] = reinterpret_cast<const uint4*>(row)[0];
    d.u[1] = reinterpret_cast<const uint4*>(row)[1];

    float acc[ACT] = {0.0f, 0.0f, 0.0f, 0.0f};
#pragma unroll
    for (int i = 0; i < 16; ++i) {
      const float s = (float)d.h[i];
#pragma unroll
      for (int a = 0; a < ACT; ++a) acc[a] += s * wout[a * H2 + lane * 16 + i];
    }
#pragma unroll
    for (int a = 0; a < ACT; ++a)
      for (int off = 16; off >= 1; off >>= 1)
        acc[a] += __shfl_xor(acc[a], off, 32);
#pragma unroll
    for (int a = 0; a < ACT; ++a) {
      memo[a] = beta * memo[a] + (acc[a] + bout[a]);
      mx[a] = fmaxf(mx[a], memo[a]);
    }
  }
  if (lane == 0) {
#pragma unroll
    for (int a = 0; a < ACT; ++a) out[(size_t)b * ACT + a] = mx[a];
  }
}

// ---------------------------------------------------------------------------
extern "C" void kernel_launch(void* const* d_in, const int* in_sizes, int n_in,
                              void* d_out, int out_size, void* d_ws, size_t ws_size,
                              hipStream_t stream) {
  const float* x     = (const float*)d_in[0];
  const float* W1    = (const float*)d_in[1];
  const float* b1    = (const float*)d_in[2];
  const float* W2    = (const float*)d_in[3];
  const float* b2    = (const float*)d_in[4];
  const float* Wout  = (const float*)d_in[5];
  const float* bout  = (const float*)d_in[6];
  const float* beta1 = (const float*)d_in[7];
  const float* beta2 = (const float*)d_in[8];
  const float* betao = (const float*)d_in[9];
  float* out = (float*)d_out;

  char* ws = (char*)d_ws;
  size_t off = 0;
  _Float16* spk1 = (_Float16*)(ws + off); off += (size_t)TSTEP * BATCH * H1 * 2;  // 64 MB
  _Float16* spk2 = (_Float16*)(ws + off); off += (size_t)TSTEP * BATCH * H2 * 2;  // 32 MB
  _Float16* xh   = (_Float16*)(ws + off); off += (size_t)BATCH * OBS * 2;         //  2 MB
  _Float16* w1h  = (_Float16*)(ws + off); off += (size_t)H1 * OBS * 2;            // .5 MB
  _Float16* w2h  = (_Float16*)(ws + off); off += (size_t)H2 * H1 * 2;             //  1 MB

  {
    int n = BATCH * OBS;
    cvt_f32_f16<<<(n + 255) / 256, 256, 0, stream>>>(x, xh, n);
    n = H1 * OBS;
    cvt_f32_f16<<<(n + 255) / 256, 256, 0, stream>>>(W1, w1h, n);
    n = H2 * H1;
    cvt_f32_f16<<<(n + 255) / 256, 256, 0, stream>>>(W2, w2h, n);
  }

  // GEMM1 + LIF1: (4096/16)*(1024/16) = 16384 wave-tiles, 8 waves / 256-thr block
  gemm1_lif1<<<(BATCH / 16) * (H1 / 16) / 8, 256, 0, stream>>>(xh, w1h, b1, beta1, spk1);

  // GEMM2 + LIF2 (2x2 tiles): (4096/32)*(512/32) = 2048 wave-tiles
  gemm2_lif2<<<(BATCH / 32) * (H2 / 32) / 8, 256, 0, stream>>>(spk1, w2h, b2, beta2, spk2);

  // Output layer: one wave per batch row
  out_layer<<<BATCH / 8, 256, 0, stream>>>(spk2, Wout, bout, betao, out);
}
